// Attention_13718125543822
// MI455X (gfx1250) — matrix operand order repaired and checked by reading, not compiled, not run on hardware
//
#include <hip/hip_runtime.h>

typedef _Float16 half_t;
typedef __attribute__((ext_vector_type(16))) _Float16 v16h;
typedef __attribute__((ext_vector_type(8)))  _Float16 v8h;
typedef __attribute__((ext_vector_type(8)))  float    v8f;
typedef __attribute__((ext_vector_type(4)))  int      v4i;

union V16 { struct { v8h lo, hi; } p; v16h v; };

__device__ __forceinline__ v8f vzero8() {
  v8f z;
  for (int i = 0; i < 8; ++i) z[i] = 0.f;
  return z;
}

#define NSP   100352L   // 32*56*56 spatial positions
#define CDIM  384
#define MQKV  1152
#define HEADS 12
#define DH    32
#define WINS  2048
#define TOKP  64        // 49 tokens padded to 64

#if __has_builtin(__builtin_amdgcn_global_load_async_to_lds_b128) && \
    __has_builtin(__builtin_amdgcn_s_wait_asynccnt)
#define USE_ASYNC_LDS 1
typedef __attribute__((address_space(1))) v4i* gvec_p;   // global int4*
typedef __attribute__((address_space(3))) v4i* lvec_p;   // LDS int4*
#else
#define USE_ASYNC_LDS 0
#endif

// Stage one 64x384 f16 weight slab (48 KB) into LDS, whole workgroup (256 thr).
// 3072 x 16B chunks, 12 per thread.
__device__ __forceinline__ void stage_weights(const half_t* __restrict__ src,
                                              half_t* __restrict__ dst) {
#if USE_ASYNC_LDS
  #pragma unroll
  for (int i = 0; i < 12; ++i) {
    const int idx = threadIdx.x + 256 * i;
    __builtin_amdgcn_global_load_async_to_lds_b128((gvec_p)(src + idx * 8),
                                                   (lvec_p)(dst + idx * 8), 0, 0);
  }
  __builtin_amdgcn_s_wait_asynccnt(0);
#else
  #pragma unroll
  for (int i = 0; i < 12; ++i) {
    const int idx = threadIdx.x + 256 * i;
    *(uint4*)(dst + idx * 8) = *(const uint4*)(src + idx * 8);
  }
#endif
  __syncthreads();
}

// ---------------------------------------------------------------------------
// fp32 -> f16 weight conversion
// ---------------------------------------------------------------------------
__global__ void convert_kernel(const float* __restrict__ in, half_t* __restrict__ out, int n) {
  int i = blockIdx.x * blockDim.x + threadIdx.x;
  if (i < n) out[i] = (half_t)in[i];
}

// ---------------------------------------------------------------------------
// x (b, c, p) fp32 -> xT (n = b*3136+p, c) f16, LDS-tiled transpose
// grid (98, 12, 32), block (32, 8)
// ---------------------------------------------------------------------------
__global__ __launch_bounds__(256) void transpose_x_kernel(const float* __restrict__ x,
                                                          half_t* __restrict__ xT) {
  __shared__ float tile[32][33];
  const int b = blockIdx.z;
  const int c0 = blockIdx.y * 32, p0 = blockIdx.x * 32;
#pragma unroll
  for (int k = 0; k < 4; ++k) {
    const int c = threadIdx.y + 8 * k;
    tile[c][threadIdx.x] = x[((long)b * CDIM + c0 + c) * 3136 + p0 + threadIdx.x];
  }
  __syncthreads();
#pragma unroll
  for (int k = 0; k < 4; ++k) {
    const int pr = threadIdx.y + 8 * k;
    xT[((long)b * 3136 + p0 + pr) * CDIM + c0 + threadIdx.x] = (half_t)tile[threadIdx.x][pr];
  }
}

// ---------------------------------------------------------------------------
// QKV GEMM: qkv[o][n] = sum_c wq[o][c] * xT[n][c].
// Block = (mblock, 8 consecutive ntiles); weights staged in LDS via async copy.
// Each wave: 64(M)x16(N), A-frags from LDS (ds_load_b128), B-frags from global.
// Epilogue scatters to windowed Q (scaled), K, and transposed V.
// grid (784, 18), block 256
// ---------------------------------------------------------------------------
__global__ __launch_bounds__(256) void qkv_gemm_kernel(
    const half_t* __restrict__ xT, const half_t* __restrict__ wq,
    half_t* __restrict__ Qb, half_t* __restrict__ Kb, half_t* __restrict__ Vt) {
  __shared__ __align__(16) half_t Wlds[64 * CDIM];  // 48 KB
  const int lane = threadIdx.x & 31;
  const int li = lane & 15, g = lane >> 4;
  const int mblock = blockIdx.y;                          // 0..17
  const int ntile  = blockIdx.x * 8 + (threadIdx.x >> 5); // 0..6271

  stage_weights(wq + (long)mblock * 64 * CDIM, Wlds);

  const int n = ntile * 16 + li;
  const int b = n / 3136, p = n % 3136;
  const int y = p / 56, x = p % 56;
  const int win = b * 64 + (y / 7) * 8 + (x / 7);
  const int tok = (y % 7) * 7 + (x % 7);

  v8f acc[4];
  for (int i = 0; i < 4; ++i) acc[i] = vzero8();

  const half_t* xcol = xT + (long)n * CDIM;
#pragma unroll
  for (int ks = 0; ks < 12; ++ks) {
    const v16h bfrag = *(const v16h*)(xcol + ks * 32 + g * 16);  // B: 16 contiguous K
    const int ka = ks * 32 + g * 16;                             // A: 16 contiguous K
#pragma unroll
    for (int mt = 0; mt < 4; ++mt) {
      const half_t* wr = Wlds + (mt * 16 + li) * CDIM + ka;
      V16 a; a.p.lo = *(const v8h*)wr; a.p.hi = *(const v8h*)(wr + 8);
      acc[mt] = __builtin_amdgcn_wmma_f32_16x16x32_f16(false, a.v, false, bfrag,
                                                       (short)0, acc[mt], false, false);
    }
  }

  const long wh = (long)win * HEADS;
#pragma unroll
  for (int mt = 0; mt < 4; ++mt) {
#pragma unroll
    for (int r = 0; r < 8; ++r) {
      const int o = mblock * 64 + mt * 16 + g * 8 + r;  // C layout: row = r + 8g
      const int part = o / CDIM, oc = o % CDIM;
      const int head = oc >> 5, d = oc & 31;
      const float v = acc[mt][r];
      const long base = wh + head;
      if (part == 0)      Qb[(base * TOKP + tok) * DH + d] = (half_t)(v * 0.17677669529663687f);
      else if (part == 1) Kb[(base * TOKP + tok) * DH + d] = (half_t)v;
      else                Vt[(base * DH + d) * TOKP + tok] = (half_t)v;  // V transposed
    }
  }
}

// ---------------------------------------------------------------------------
// Windowed attention: one wave per (window, head). 4 waves / block.
// ---------------------------------------------------------------------------
__global__ __launch_bounds__(128) void attn_win_kernel(
    const half_t* __restrict__ Qb, const half_t* __restrict__ Kb,
    const half_t* __restrict__ Vt, const float* __restrict__ bias_table,
    half_t* __restrict__ attnout) {
  __shared__ __align__(128) half_t Ps[4][TOKP * TOKP];
  const int wv = threadIdx.x >> 5;
  const int lane = threadIdx.x & 31;
  const int li = lane & 15, g = lane >> 4;
  const int task = blockIdx.x * 4 + wv;
  const int win = task / HEADS, head = task % HEADS;

  const half_t* Qw = Qb + ((long)win * HEADS + head) * (TOKP * DH);
  const half_t* Kw = Kb + ((long)win * HEADS + head) * (TOKP * DH);
  const half_t* Vw = Vt + ((long)win * HEADS + head) * (DH * TOKP);

  // ---- S = (q*scale) . k^T ; K-dim = d = 32, single WMMA per tile ----
  v16h qa[4];
#pragma unroll
  for (int mt = 0; mt < 4; ++mt) {
    const half_t* qr = Qw + (mt * 16 + li) * DH + g * 16;
    V16 a; a.p.lo = *(const v8h*)qr; a.p.hi = *(const v8h*)(qr + 8);
    qa[mt] = a.v;
  }
  v8f s[4][4];
#pragma unroll
  for (int nt = 0; nt < 4; ++nt) {
    const v16h kb = *(const v16h*)(Kw + (nt * 16 + li) * DH + g * 16);  // K^T columns
#pragma unroll
    for (int mt = 0; mt < 4; ++mt) {
      v8f z = vzero8();
      s[mt][nt] = __builtin_amdgcn_wmma_f32_16x16x32_f16(false, qa[mt], false, kb,
                                                         (short)0, z, false, false);
    }
  }

  // ---- relative-position bias + pad mask + softmax in C-fragment layout ----
#pragma unroll
  for (int mt = 0; mt < 4; ++mt) {
    float rmax[8], rsum[8];
#pragma unroll
    for (int r = 0; r < 8; ++r) rmax[r] = -3.0e38f;
#pragma unroll
    for (int nt = 0; nt < 4; ++nt) {
      const int nk = nt * 16 + li;  // key token
#pragma unroll
      for (int r = 0; r < 8; ++r) {
        const int mq = mt * 16 + g * 8 + r;  // query token
        float val = s[mt][nt][r];
        if (nk < 49) {
          if (mq < 49) {
            const int dy = mq / 7 - nk / 7 + 6;
            const int dx = mq % 7 - nk % 7 + 6;
            val += bias_table[(dy * 13 + dx) * HEADS + head];
          }
        } else {
          val = -3.0e38f;  // mask padded key columns
        }
        s[mt][nt][r] = val;
        rmax[r] = fmaxf(rmax[r], val);
      }
    }
#pragma unroll
    for (int r = 0; r < 8; ++r) {  // butterfly within 16-lane halves (C layout)
      float m = rmax[r];
      m = fmaxf(m, __shfl_xor(m, 1, 32));
      m = fmaxf(m, __shfl_xor(m, 2, 32));
      m = fmaxf(m, __shfl_xor(m, 4, 32));
      m = fmaxf(m, __shfl_xor(m, 8, 32));
      rmax[r] = m;
      rsum[r] = 0.f;
    }
#pragma unroll
    for (int nt = 0; nt < 4; ++nt)
#pragma unroll
      for (int r = 0; r < 8; ++r) {
        const float e = __expf(s[mt][nt][r] - rmax[r]);
        s[mt][nt][r] = e;
        rsum[r] += e;
      }
#pragma unroll
    for (int r = 0; r < 8; ++r) {
      float t = rsum[r];
      t += __shfl_xor(t, 1, 32);
      t += __shfl_xor(t, 2, 32);
      t += __shfl_xor(t, 4, 32);
      t += __shfl_xor(t, 8, 32);
      rsum[r] = 1.0f / t;
    }
#pragma unroll
    for (int nt = 0; nt < 4; ++nt)
#pragma unroll
      for (int r = 0; r < 8; ++r)
        Ps[wv][(mt * 16 + g * 8 + r) * TOKP + nt * 16 + li] = (half_t)(s[mt][nt][r] * rsum[r]);
  }

  __syncthreads();  // Ps written/read by different lanes

  // ---- O = P . V ; K-dim = 64 padded tokens (V pad is zeroed) ----
  v8f oacc[4][2];
  for (int mt = 0; mt < 4; ++mt)
    for (int nd = 0; nd < 2; ++nd) oacc[mt][nd] = vzero8();
#pragma unroll
  for (int ks = 0; ks < 2; ++ks) {
    v16h vb[2];
#pragma unroll
    for (int nd = 0; nd < 2; ++nd)
      vb[nd] = *(const v16h*)(Vw + (nd * 16 + li) * TOKP + ks * 32 + g * 16);
#pragma unroll
    for (int mt = 0; mt < 4; ++mt) {
      const half_t* pr = &Ps[wv][(mt * 16 + li) * TOKP + ks * 32 + g * 16];
      V16 a; a.p.lo = *(const v8h*)pr; a.p.hi = *(const v8h*)(pr + 8);
#pragma unroll
      for (int nd = 0; nd < 2; ++nd)
        oacc[mt][nd] = __builtin_amdgcn_wmma_f32_16x16x32_f16(false, a.v, false, vb[nd],
                                                              (short)0, oacc[mt][nd], false, false);
    }
  }

  // ---- window-reverse scatter into (n_spatial, c) f16 ----
  const int wb = win / 64, rem = win % 64, wy = rem / 8, wx = rem % 8;
#pragma unroll
  for (int mt = 0; mt < 4; ++mt) {
#pragma unroll
    for (int r = 0; r < 8; ++r) {
      const int tk = mt * 16 + g * 8 + r;
      if (tk < 49) {
        const int yy = wy * 7 + tk / 7, xx = wx * 7 + tk % 7;
        const long nsp = (long)wb * 3136 + yy * 56 + xx;
#pragma unroll
        for (int nd = 0; nd < 2; ++nd)
          attnout[nsp * CDIM + head * DH + nd * 16 + li] = (half_t)oacc[mt][nd][r];
      }
    }
  }
}

// ---------------------------------------------------------------------------
// Output projection GEMM with bias, fp32 result in (b, o, h, w)
// grid (784, 6), block 256; weights staged in LDS like qkv_gemm.
// ---------------------------------------------------------------------------
__global__ __launch_bounds__(256) void proj_gemm_kernel(
    const half_t* __restrict__ att, const half_t* __restrict__ wp,
    const float* __restrict__ bproj, float* __restrict__ out) {
  __shared__ __align__(16) half_t Wlds[64 * CDIM];  // 48 KB
  const int lane = threadIdx.x & 31;
  const int li = lane & 15, g = lane >> 4;
  const int mblock = blockIdx.y;                          // 0..5
  const int ntile  = blockIdx.x * 8 + (threadIdx.x >> 5);

  stage_weights(wp + (long)mblock * 64 * CDIM, Wlds);

  const int n = ntile * 16 + li;
  const int b = n / 3136, p = n % 3136;

  v8f acc[4];
  for (int i = 0; i < 4; ++i) acc[i] = vzero8();

  const half_t* acol = att + (long)n * CDIM;
#pragma unroll
  for (int ks = 0; ks < 12; ++ks) {
    const v16h bfrag = *(const v16h*)(acol + ks * 32 + g * 16);
    const int ka = ks * 32 + g * 16;
#pragma unroll
    for (int mt = 0; mt < 4; ++mt) {
      const half_t* wr = Wlds + (mt * 16 + li) * CDIM + ka;
      V16 a; a.p.lo = *(const v8h*)wr; a.p.hi = *(const v8h*)(wr + 8);
      acc[mt] = __builtin_amdgcn_wmma_f32_16x16x32_f16(false, a.v, false, bfrag,
                                                       (short)0, acc[mt], false, false);
    }
  }

#pragma unroll
  for (int mt = 0; mt < 4; ++mt)
#pragma unroll
    for (int r = 0; r < 8; ++r) {
      const int o = mblock * 64 + mt * 16 + g * 8 + r;
      out[((long)b * CDIM + o) * 3136 + p] = acc[mt][r] + bproj[o];
    }
}

// ---------------------------------------------------------------------------
extern "C" void kernel_launch(void* const* d_in, const int* in_sizes, int n_in,
                              void* d_out, int out_size, void* d_ws, size_t ws_size,
                              hipStream_t stream) {
  const float* x      = (const float*)d_in[0];
  const float* w_qkv  = (const float*)d_in[1];
  const float* btab   = (const float*)d_in[2];
  const float* w_proj = (const float*)d_in[3];
  const float* b_proj = (const float*)d_in[4];
  float* out = (float*)d_out;

  char* ws = (char*)d_ws;
  const long XT_B = NSP * CDIM * 2;                      // 77,070,336
  const long WQ_B = (long)MQKV * CDIM * 2;               //    884,736
  const long WP_B = (long)CDIM * CDIM * 2;               //    294,912
  const long QK_B = (long)WINS * HEADS * TOKP * DH * 2;  // 100,663,296

  half_t* xT  = (half_t*)(ws);
  half_t* wqb = (half_t*)(ws + XT_B);
  half_t* wpb = (half_t*)(ws + XT_B + WQ_B);
  half_t* Qb  = (half_t*)(ws + XT_B + WQ_B + WP_B);
  half_t* Kb  = (half_t*)(ws + XT_B + WQ_B + WP_B + QK_B);
  half_t* Vt  = (half_t*)(ws + XT_B + WQ_B + WP_B + 2 * QK_B);
  half_t* attnout = xT;  // xT fully consumed by qkv_gemm before attn writes here

  convert_kernel<<<(MQKV * CDIM + 255) / 256, 256, 0, stream>>>(w_qkv, wqb, MQKV * CDIM);
  convert_kernel<<<(CDIM * CDIM + 255) / 256, 256, 0, stream>>>(w_proj, wpb, CDIM * CDIM);
  transpose_x_kernel<<<dim3(98, 12, 32), dim3(32, 8), 0, stream>>>(x, xT);
  (void)hipMemsetAsync(Vt, 0, (size_t)QK_B, stream);  // zero V pad so P·V pad adds exactly 0

  qkv_gemm_kernel<<<dim3(784, 18), 256, 0, stream>>>(xT, wqb, Qb, Kb, Vt);
  attn_win_kernel<<<WINS * HEADS / 4, 128, 0, stream>>>(Qb, Kb, Vt, btab, attnout);
  proj_gemm_kernel<<<dim3(784, 6), 256, 0, stream>>>(attnout, wpb, b_proj, out);
}
